// Discriminator_42898133352735
// MI455X (gfx1250) — compile-verified, matrix operand order checked by reading, not run on hardware
//
#include <hip/hip_runtime.h>
#include <hip/hip_bf16.h>

// ---------------------------------------------------------------------------
// CDNA5 (gfx1250) implementation. wave32; WMMA 16x16x32 bf16 (f32 accum);
// async global->LDS staging (ASYNCcnt) in the all-pairs kernel.
// ---------------------------------------------------------------------------

typedef __attribute__((ext_vector_type(16))) __bf16 v16bf;
typedef __attribute__((ext_vector_type(8)))  float  v8f;

__device__ __forceinline__ float lrelu(float v) { return v > 0.0f ? v : 0.2f * v; }

// ---- WMMA fragment builders (ISA 7.12.2 16-bit layouts) -------------------
// A (16xK): lane 0-15 -> M=lane, K in {kb..kb+7, kb+16..kb+23}, kb = k0 + 8*(lane>=16)
__device__ __forceinline__ v16bf load_a_global(const float* __restrict__ A, int lda,
                                               int row0, int k0, int lane) {
  const int m  = row0 + (lane & 15);
  const int kb = k0 + ((lane >> 4) << 3);
  const float* p = A + (size_t)m * lda + kb;
  v16bf a;
#pragma unroll
  for (int e = 0; e < 8; ++e) a[e] = (__bf16)p[e];
#pragma unroll
  for (int e = 0; e < 8; ++e) a[e + 8] = (__bf16)p[e + 16];
  return a;
}

// B (Kx16, row-major W[k*ldb + n]); symmetric lane=N layout.
__device__ __forceinline__ v16bf load_b_global(const float* __restrict__ W, int ldb,
                                               int col0, int k0, int nvalid, int lane) {
  const int n  = col0 + (lane & 15);
  const int kb = k0 + ((lane >> 4) << 3);
  v16bf b;
#pragma unroll
  for (int e = 0; e < 16; ++e) {
    const int k = kb + (e < 8 ? e : e + 8);
    float w = 0.0f;
    if (n < nvalid) w = W[(size_t)k * ldb + n];
    b[e] = (__bf16)w;
  }
  return b;
}

// A fragment sourced from a 16x64 f32 LDS tile.
__device__ __forceinline__ v16bf load_a_lds(const float (*buf)[64], int k0, int lane) {
  const int m  = lane & 15;
  const int kb = k0 + ((lane >> 4) << 3);
  v16bf a;
#pragma unroll
  for (int e = 0; e < 16; ++e) {
    const int k = kb + (e < 8 ? e : e + 8);
    a[e] = (__bf16)buf[m][k];
  }
  return a;
}

// ---- Generic GEMM: C = act(A[1024,K] @ W[K,N] + b)
// One wave owns a 16x64 output strip: 4 accumulators share each A fragment,
// so A is loaded/converted once per k-step and 4 v_wmma issue back-to-back.
template <int ACT>
__global__ __launch_bounds__(32) void wmma_gemm_bias(const float* __restrict__ A,
                                                     const float* __restrict__ W,
                                                     const float* __restrict__ bias,
                                                     float* __restrict__ C,
                                                     int K, int N) {
  const int lane = threadIdx.x;
  const int col0 = blockIdx.x * 64;
  const int row0 = blockIdx.y * 16;
  v8f acc[4] = {v8f{}, v8f{}, v8f{}, v8f{}};
  for (int k0 = 0; k0 < K; k0 += 32) {
    if (k0 + 32 < K) __builtin_prefetch(W + (size_t)(k0 + 32) * N + col0, 0, 1);
    v16bf a = load_a_global(A, K, row0, k0, lane);
#pragma unroll
    for (int nt = 0; nt < 4; ++nt) {
      v16bf b = load_b_global(W, N, col0 + nt * 16, k0, N, lane);
      acc[nt] = __builtin_amdgcn_wmma_f32_16x16x32_bf16(false, a, false, b, (short)0,
                                                        acc[nt], false, false);
    }
  }
  // D layout: lane -> N = lane&15; VGPR r -> M = r + 8*(lane>=16)
  const int rbase = (lane >> 4) << 3;
#pragma unroll
  for (int nt = 0; nt < 4; ++nt) {
    const int n    = col0 + nt * 16 + (lane & 15);
    const float bn = bias[n];
#pragma unroll
    for (int r = 0; r < 8; ++r) {
      float v = acc[nt][r] + bn;
      if (ACT) v = lrelu(v);
      C[(size_t)(row0 + rbase + r) * N + n] = v;
    }
  }
}

// ---- BatchNorm statistics: one block per feature column (128 cols, 1024 rows)
__global__ __launch_bounds__(256) void bn_stats(const float* __restrict__ H2,
                                                float* __restrict__ mu,
                                                float* __restrict__ var) {
  __shared__ float s1[256];
  __shared__ float s2[256];
  const int c = blockIdx.x;
  const int t = threadIdx.x;
  float s = 0.0f, ss = 0.0f;
  for (int r = t; r < 1024; r += 256) {
    float v = H2[(size_t)r * 128 + c];
    s += v; ss += v * v;
  }
  s1[t] = s; s2[t] = ss;
  __syncthreads();
  for (int w = 128; w > 0; w >>= 1) {
    if (t < w) { s1[t] += s1[t + w]; s2[t] += s2[t + w]; }
    __syncthreads();
  }
  if (t == 0) {
    float m = s1[0] * (1.0f / 1024.0f);
    mu[c]  = m;
    var[c] = s2[0] * (1.0f / 1024.0f) - m * m;   // biased variance (torch BN)
  }
}

// ---- Fused tail per 16-row tile: BN->lrelu -> GEMM3 -> lrelu ->
//      attention residual (Wv,Wo) -> M = H5 @ T  (all WMMA, LDS re-layout)
__global__ __launch_bounds__(32) void fused_tail(
    const float* __restrict__ H2, const float* __restrict__ mu, const float* __restrict__ var,
    const float* __restrict__ gamma, const float* __restrict__ beta,
    const float* __restrict__ W3, const float* __restrict__ b3,
    const float* __restrict__ Wv, const float* __restrict__ bv,
    const float* __restrict__ Wo, const float* __restrict__ bo,
    const float* __restrict__ T, float* __restrict__ H5g, float* __restrict__ Mout) {
  __shared__ float h4[16][64];
  __shared__ float tmp[16][64];
  __shared__ float h5[16][64];
  const int lane  = threadIdx.x;
  const int row0  = blockIdx.x * 16;
  const int nloc  = lane & 15;
  const int rbase = (lane >> 4) << 3;

  // Layer3: lrelu( BN(H2) @ W3 + b3 ), K=128, N=64
  for (int nt = 0; nt < 4; ++nt) {
    v8f c = {};
    for (int k0 = 0; k0 < 128; k0 += 32) {
      // A fragment: BN + lrelu applied on the fly
      const int m  = row0 + (lane & 15);
      const int kb = k0 + ((lane >> 4) << 3);
      v16bf a;
#pragma unroll
      for (int e = 0; e < 16; ++e) {
        const int k = kb + (e < 8 ? e : e + 8);
        float v = H2[(size_t)m * 128 + k];
        v = (v - mu[k]) * rsqrtf(var[k] + 1e-5f) * gamma[k] + beta[k];
        a[e] = (__bf16)lrelu(v);
      }
      v16bf b = load_b_global(W3, 64, nt * 16, k0, 64, lane);
      c = __builtin_amdgcn_wmma_f32_16x16x32_bf16(false, a, false, b, (short)0, c, false, false);
    }
    const int n    = nt * 16 + nloc;
    const float bn = b3[n];
#pragma unroll
    for (int r = 0; r < 8; ++r) h4[rbase + r][n] = lrelu(c[r] + bn);
  }
  __syncthreads();

  // V = H4 @ Wv + bv   (K=64)
  for (int nt = 0; nt < 4; ++nt) {
    v8f c = {};
    for (int k0 = 0; k0 < 64; k0 += 32) {
      v16bf a = load_a_lds(h4, k0, lane);
      v16bf b = load_b_global(Wv, 64, nt * 16, k0, 64, lane);
      c = __builtin_amdgcn_wmma_f32_16x16x32_bf16(false, a, false, b, (short)0, c, false, false);
    }
    const int n    = nt * 16 + nloc;
    const float bn = bv[n];
#pragma unroll
    for (int r = 0; r < 8; ++r) tmp[rbase + r][n] = c[r] + bn;
  }
  __syncthreads();

  // H5 = H4 + V @ Wo + bo
  for (int nt = 0; nt < 4; ++nt) {
    v8f c = {};
    for (int k0 = 0; k0 < 64; k0 += 32) {
      v16bf a = load_a_lds(tmp, k0, lane);
      v16bf b = load_b_global(Wo, 64, nt * 16, k0, 64, lane);
      c = __builtin_amdgcn_wmma_f32_16x16x32_bf16(false, a, false, b, (short)0, c, false, false);
    }
    const int n    = nt * 16 + nloc;
    const float bn = bo[n];
#pragma unroll
    for (int r = 0; r < 8; ++r) {
      float v = c[r] + bn + h4[rbase + r][n];
      h5[rbase + r][n] = v;
      H5g[(size_t)(row0 + rbase + r) * 64 + n] = v;
    }
  }
  __syncthreads();

  // M = H5 @ T[64,250]  (N padded 250 -> 256 in Mout)
  for (int nt = 0; nt < 16; ++nt) {
    v8f c = {};
    for (int k0 = 0; k0 < 64; k0 += 32) {
      v16bf a = load_a_lds(h5, k0, lane);
      v16bf b = load_b_global(T, 250, nt * 16, k0, 250, lane);
      c = __builtin_amdgcn_wmma_f32_16x16x32_bf16(false, a, false, b, (short)0, c, false, false);
    }
    const int n = nt * 16 + nloc;
    if (n < 250) {
#pragma unroll
      for (int r = 0; r < 8; ++r)
        Mout[(size_t)(row0 + rbase + r) * 256 + n] = c[r];
    }
  }
}

// ---- All-pairs minibatch discrimination: o[j,f] = sum_i exp(-||M_i,f - M_j,f||_1) - 1
// Block: 256 threads = 5 j-rows x 50 features. M rows staged through LDS via
// gfx1250 async global->LDS copies (ASYNCcnt), 16B per lane-op.
__global__ __launch_bounds__(256) void pairwise(const float* __restrict__ Mb,
                                                float* __restrict__ ob) {
  __shared__ float smem[8 * 256];
  const int t  = threadIdx.x;
  const int jl = t / 50;
  const int f  = t % 50;
  const int j  = blockIdx.x * 5 + jl;
  const bool active = (jl < 5) && (j < 1024);
  float mj[5];
  if (active) {
#pragma unroll
    for (int kk = 0; kk < 5; ++kk) mj[kk] = Mb[(size_t)j * 256 + f * 5 + kk];
  }
  float acc = 0.0f;
  for (int i0 = 0; i0 < 1024; i0 += 8) {
    // async-stage 8 rows x 256 floats (8 KB) into LDS: 512 x b128, 2 per thread
#pragma unroll
    for (int rep = 0; rep < 2; ++rep) {
      const int idx4 = t + rep * 256;                 // 16-byte chunk index
      unsigned lds_addr = (unsigned)(uintptr_t)(&smem[idx4 * 4]);
      const float* gp   = Mb + (size_t)i0 * 256 + idx4 * 4;
      asm volatile("global_load_async_to_lds_b128 %0, %1, off"
                   :: "v"(lds_addr), "v"(gp)
                   : "memory");
    }
    asm volatile("s_wait_asynccnt 0x0" ::: "memory");
    __syncthreads();
    if (active) {
#pragma unroll
      for (int ri = 0; ri < 8; ++ri) {
        const float* mi = &smem[ri * 256 + f * 5];
        float s = 0.0f;
#pragma unroll
        for (int kk = 0; kk < 5; ++kk) s += fabsf(mi[kk] - mj[kk]);
        acc += __expf(-s);
      }
    }
    __syncthreads();
  }
  if (active) ob[(size_t)j * 50 + f] = acc - 1.0f;  // remove self term exp(0)
}

// ---- Final score: [h5 | o] @ Ws + bs
__global__ __launch_bounds__(256) void finalize(const float* __restrict__ H5,
                                                const float* __restrict__ ob,
                                                const float* __restrict__ Ws,
                                                const float* __restrict__ bs,
                                                float* __restrict__ out) {
  const int j = blockIdx.x * 256 + threadIdx.x;
  if (j >= 1024) return;
  float s = bs[0];
#pragma unroll 4
  for (int c = 0; c < 64; ++c) s += H5[(size_t)j * 64 + c] * Ws[c];
#pragma unroll 5
  for (int f = 0; f < 50; ++f) s += ob[(size_t)j * 50 + f] * Ws[64 + f];
  out[j] = s;
}

extern "C" void kernel_launch(void* const* d_in, const int* in_sizes, int n_in,
                              void* d_out, int out_size, void* d_ws, size_t ws_size,
                              hipStream_t stream) {
  (void)in_sizes; (void)n_in; (void)out_size; (void)ws_size;
  const float* x     = (const float*)d_in[0];
  const float* W1    = (const float*)d_in[1];
  const float* b1    = (const float*)d_in[2];
  const float* W2    = (const float*)d_in[3];
  const float* b2    = (const float*)d_in[4];
  const float* gamma = (const float*)d_in[5];
  const float* beta  = (const float*)d_in[6];
  const float* W3    = (const float*)d_in[7];
  const float* b3    = (const float*)d_in[8];
  const float* Wv    = (const float*)d_in[9];
  const float* bv    = (const float*)d_in[10];
  const float* Wo    = (const float*)d_in[11];
  const float* bo    = (const float*)d_in[12];
  const float* T     = (const float*)d_in[13];
  const float* Ws    = (const float*)d_in[14];
  const float* bs    = (const float*)d_in[15];

  float* ws  = (float*)d_ws;
  float* H1  = ws;                   // 1024*256
  float* H2  = H1  + 1024 * 256;     // 1024*128
  float* muv = H2  + 1024 * 128;     // 128
  float* var = muv + 128;            // 128
  float* H5  = var + 128;            // 1024*64
  float* Mb  = H5  + 1024 * 64;      // 1024*256 (250 valid cols, padded)
  float* ob  = Mb  + 1024 * 256;     // 1024*50

  // H1 = lrelu(x @ W1 + b1)                       [1024,256]
  wmma_gemm_bias<1><<<dim3(4, 64), 32, 0, stream>>>(x, W1, b1, H1, 128, 256);
  // H2 = H1 @ W2 + b2 (pre-BN)                    [1024,128]
  wmma_gemm_bias<0><<<dim3(2, 64), 32, 0, stream>>>(H1, W2, b2, H2, 256, 128);
  // batch statistics
  bn_stats<<<128, 256, 0, stream>>>(H2, muv, var);
  // BN->lrelu->GEMM3->lrelu->attention residual->M projection
  fused_tail<<<64, 32, 0, stream>>>(H2, muv, var, gamma, beta, W3, b3,
                                    Wv, bv, Wo, bo, T, H5, Mb);
  // minibatch discrimination all-pairs
  pairwise<<<205, 256, 0, stream>>>(Mb, ob);
  // score = [H5 | o] @ Ws + bs
  finalize<<<4, 256, 0, stream>>>(H5, ob, Ws, bs, (float*)d_out);
}
